// PAM_6923487281401
// MI455X (gfx1250) — matrix operand order repaired and checked by reading, not compiled, not run on hardware
//
#include <hip/hip_runtime.h>

// ---------------------------------------------------------------------------
// PAM attention for MI455X (gfx1250, wave32, WMMA).
//   Q = Wq x, K = Wk x, V = Wv x        (tiny, VALU kernel, bf16 outputs)
//   QK[b,q,k] = sum_d Q[d,q] K[d,k]     (must be fully materialized: 268 MB)
//   out = alpha * softmax(QK) V^T + x   (fused flash-style, bf16 WMMA)
// HBM floor ~ 276 MB => ~12us @ 23.3 TB/s; 10 GFLOP is negligible for WMMA.
// QK stores are non-temporal (write-once stream); K/V stay L2-resident.
// ---------------------------------------------------------------------------

typedef __attribute__((ext_vector_type(16))) __bf16 v16bf;
typedef __attribute__((ext_vector_type(8)))  float  v8f;
typedef __attribute__((ext_vector_type(4)))  float  vf4;

#define NN 4096   // H*W
#define NC 64     // channels
#define ND 8      // qk dim
#define NB 4      // batch
#define PART_F 1088  // floats per (tile,split) partial: 1024 acc + 32 m + 32 l
static constexpr float L2E = 1.44269504f;

// Raw v_exp_f32 (no subnormal fixup: softmax tails below 2^-126 are noise).
#if __has_builtin(__builtin_amdgcn_exp2f)
#define EXP2(x) __builtin_amdgcn_exp2f(x)
#else
#define EXP2(x) exp2f(x)
#endif

__device__ __forceinline__ __bf16 f2bf(float f) {       // RNE (cold paths)
  unsigned u = __builtin_bit_cast(unsigned, f);
  unsigned r = u + 0x7fffu + ((u >> 16) & 1u);
  unsigned short h = (unsigned short)(r >> 16);
  return __builtin_bit_cast(__bf16, h);
}

// Pack two f32 into bf16x2 by truncation: one v_perm_b32 (hot path).
__device__ __forceinline__ unsigned pk_bf16(float hi, float lo) {
  unsigned uh = __builtin_bit_cast(unsigned, hi);
  unsigned ul = __builtin_bit_cast(unsigned, lo);
#if __has_builtin(__builtin_amdgcn_perm)
  return __builtin_amdgcn_perm(uh, ul, 0x07060302u);
#else
  return (uh & 0xffff0000u) | (ul >> 16);
#endif
}

union FragU { v16bf v; uint4 q4[2]; unsigned u[8]; };
union V8U   { v8f v; vf4 f4[2]; float f[8]; };

// ---------------------------------------------------------------------------
// Kernel 1: 1x1-conv projections -> Qt[N,8], Kt[N,8], V[64,N]  (bf16).
// Also zeroes the 256B pad used for branchless zero-padded WMMA fragments.
// ---------------------------------------------------------------------------
__global__ __launch_bounds__(256) void pam_proj(
    const float* __restrict__ x,
    const float* __restrict__ Wq, const float* __restrict__ bq,
    const float* __restrict__ Wk, const float* __restrict__ bk,
    const float* __restrict__ Wv, const float* __restrict__ bv,
    __bf16* __restrict__ Qt, __bf16* __restrict__ Kt, __bf16* __restrict__ Vm,
    unsigned* __restrict__ zeropad) {
  if (blockIdx.x == 0 && blockIdx.y == 0 && threadIdx.x < 64)
    zeropad[threadIdx.x] = 0u;
  int n = blockIdx.x * 256 + threadIdx.x;
  int b = blockIdx.y;
  const float* xb = x + (size_t)b * NC * NN + n;
  float xc[NC];
#pragma unroll
  for (int c = 0; c < NC; ++c) xc[c] = xb[(size_t)c * NN];
#pragma unroll
  for (int d = 0; d < ND; ++d) {
    float q = bq[d], k = bk[d];
#pragma unroll
    for (int c = 0; c < NC; ++c) {
      q = fmaf(Wq[d * NC + c], xc[c], q);
      k = fmaf(Wk[d * NC + c], xc[c], k);
    }
    Qt[((size_t)b * NN + n) * ND + d] = f2bf(q);
    Kt[((size_t)b * NN + n) * ND + d] = f2bf(k);
  }
  for (int co = 0; co < NC; ++co) {          // Wv rows are scalar-uniform
    float s = bv[co];
#pragma unroll
    for (int ci = 0; ci < NC; ++ci) s = fmaf(Wv[co * NC + ci], xc[ci], s);
    Vm[((size_t)b * NC + co) * NN + n] = f2bf(s);
  }
}

// ---------------------------------------------------------------------------
// Kernel 2: fused flash attention per 16-query tile (1 wave / tile / split).
//  S^T tile = WMMA(A=Kt[16k x 32d(pad)], B=Q[32d x 16q])  -> lane owns one q
//  column, 16 k-values in-lane => online softmax is VALU + one shfl_xor(16).
//  Zero-padding via a zeroed ws region: lanes 16-31 read zeros (stride 0),
//  so the inner loop is completely branchless (no EXEC save/restore).
// ---------------------------------------------------------------------------
__global__ __launch_bounds__(32) void pam_attn(
    const __bf16* __restrict__ Qt, const __bf16* __restrict__ Kt,
    const __bf16* __restrict__ Vm, const __bf16* __restrict__ zeropad,
    const float* __restrict__ x, const float* __restrict__ alphap,
    float* __restrict__ qk, float* __restrict__ out, float* __restrict__ part,
    int nsplit, int final_out) {
  int wid   = blockIdx.x;
  int split = wid % nsplit;
  int tile  = wid / nsplit;          // 0..1023
  int b     = tile >> 8;
  int qbase = (tile & 255) << 4;
  int lane  = threadIdx.x;
  int lq    = lane & 15;
  int g     = lane >> 4;

  const uint4 Z = make_uint4(0, 0, 0, 0);
  const __bf16* Ktb  = Kt + (size_t)b * NN * ND;
  const __bf16* Vb   = Vm + (size_t)b * NC * NN;
  float*        qrow = qk + ((size_t)b * NN + (qbase + lq)) * NN;

  int klen = NN / nsplit;
  int k0 = split * klen, kend = k0 + klen;

  // Q B-fragment [32d x 16q], d zero-padded to 32. Loop-invariant, branchless.
  FragU fq;
  {
    const __bf16* qsrc =
        g ? zeropad : (Qt + ((size_t)b * NN + qbase + lq) * ND);
    fq.q4[0] = *(const uint4*)qsrc;
    fq.q4[1] = Z;
  }
  // Per-lane K-fragment stream: lanes 16-31 park on the zero pad (stride 0).
  const __bf16* kp0 = g ? zeropad : (Ktb + (size_t)(k0 + lq) * ND);
  const __bf16* kp1 = g ? zeropad : (Ktb + (size_t)(k0 + 16 + lq) * ND);
  const size_t  kadv = g ? 0 : (size_t)32 * ND;

  FragU fa0, fa1;
  fa0.q4[1] = Z;                     // padded halves: loop-invariant zeros
  fa1.q4[1] = Z;

  v8f accs[4] = {};                  // QKV tile [64c x 16q] in 4 C/D frags
  float m = -1e30f, l = 0.f;

  for (int kc = k0; kc < kend; kc += 32) {
    __builtin_prefetch(kp0 + kadv, 0, 1);   // gfx1250 global_prefetch_b8

    fa0.q4[0] = *(const uint4*)kp0;
    fa1.q4[0] = *(const uint4*)kp1;
    kp0 += kadv;
    kp1 += kadv;

    // --- S^T = K^T Q : lane L, reg r -> (k = kc + 16t + 8g + r, q = lq) ---
    v8f zc = {};
    V8U s0, s1;
    s0.v = __builtin_amdgcn_wmma_f32_16x16x32_bf16(false, fa0.v, false, fq.v,
                                                   (short)0, zc, false, false);
    s1.v = __builtin_amdgcn_wmma_f32_16x16x32_bf16(false, fa1.v, false, fq.v,
                                                   (short)0, zc, false, false);

    // --- stream raw QK tile: 8 consecutive k per lane => 2x b128 NT each ---
    __builtin_nontemporal_store(s0.f4[0], (vf4*)(qrow + kc + 8 * g));
    __builtin_nontemporal_store(s0.f4[1], (vf4*)(qrow + kc + 8 * g + 4));
    __builtin_nontemporal_store(s1.f4[0], (vf4*)(qrow + kc + 16 + 8 * g));
    __builtin_nontemporal_store(s1.f4[1], (vf4*)(qrow + kc + 16 + 8 * g + 4));

    // --- online softmax: tree max (depth 4) + one cross-half sync ---
    float mx[8];
#pragma unroll
    for (int i = 0; i < 8; ++i) mx[i] = fmaxf(s0.f[i], s1.f[i]);
#pragma unroll
    for (int i = 0; i < 4; ++i) mx[i] = fmaxf(mx[i], mx[i + 4]);
    float cm = fmaxf(fmaxf(mx[0], mx[1]), fmaxf(mx[2], mx[3]));
    cm = fmaxf(cm, __shfl_xor(cm, 16, 32));
    float mn = fmaxf(m, cm);
    float scale = EXP2((m - mn) * L2E);
    m = mn;
#pragma unroll
    for (int i = 0; i < 8; ++i) s0.f[i] = EXP2((s0.f[i] - mn) * L2E);
#pragma unroll
    for (int i = 0; i < 8; ++i) s1.f[i] = EXP2((s1.f[i] - mn) * L2E);
    float sm[8];
#pragma unroll
    for (int i = 0; i < 8; ++i) sm[i] = s0.f[i] + s1.f[i];
#pragma unroll
    for (int i = 0; i < 4; ++i) sm[i] += sm[i + 4];
    float csum = (sm[0] + sm[1]) + (sm[2] + sm[3]);
    l = fmaf(l, scale, csum);
#pragma unroll
    for (int cs = 0; cs < 4; ++cs)
#pragma unroll
      for (int i = 0; i < 8; ++i) accs[cs][i] = accs[cs][i] * scale;

    // --- build P^T B-fragment [32k x 16q]: lane g needs k=16g..16g+15 ---
    float pv[16];
#pragma unroll
    for (int j = 0; j < 8; ++j) {
      float x0 = __shfl_xor(s0.f[j], 16, 32);   // partner's tile-0 row j
      float x1 = __shfl_xor(s1.f[j], 16, 32);   // partner's tile-1 row j
      pv[j]     = g ? x1 : s0.f[j];             // k = 16g + j
      pv[8 + j] = g ? s1.f[j] : x0;             // k = 16g + 8 + j
    }
    FragU fb;
#pragma unroll
    for (int j = 0; j < 8; ++j)                 // one v_perm_b32 per pair
      fb.u[j] = pk_bf16(pv[2 * j + 1], pv[2 * j]);

    // --- acc += V[16c x 32k] * P^T, four 16-row c sub-tiles ---
#pragma unroll
    for (int cs = 0; cs < 4; ++cs) {
      const __bf16* vp = Vb + (size_t)(cs * 16 + lq) * NN + kc + 16 * g;
      __builtin_prefetch(vp + 32, 0, 1);
      FragU fv;
      fv.q4[0] = *(const uint4*)(vp);        // k = kc+16g .. +7
      fv.q4[1] = *(const uint4*)(vp + 8);    // k = kc+16g+8 .. +7
      accs[cs] = __builtin_amdgcn_wmma_f32_16x16x32_bf16(
          false, fv.v, false, fb.v, (short)0, accs[cs], false, false);
    }
  }

  if (final_out) {  // single split: normalize + residual directly
    float lt = l + __shfl_xor(l, 16, 32);
    float inv = (*alphap) / lt;
#pragma unroll
    for (int cs = 0; cs < 4; ++cs)
#pragma unroll
      for (int r = 0; r < 8; ++r) {
        int c = cs * 16 + r + 8 * g;
        size_t idx = ((size_t)b * NC + c) * NN + qbase + lq;
        out[idx] = fmaf(accs[cs][r], inv, x[idx]);
      }
  } else {          // split-K: spill (acc, m, l) partials
    float* pb = part + (size_t)(tile * nsplit + split) * PART_F;
#pragma unroll
    for (int cs = 0; cs < 4; ++cs)
#pragma unroll
      for (int r = 0; r < 8; ++r)
        pb[(cs * 8 + r) * 32 + lane] = accs[cs][r];
    pb[1024 + lane] = m;
    pb[1056 + lane] = l;
  }
}

// ---------------------------------------------------------------------------
// Kernel 3: flash split-K combine + residual (only when nsplit > 1).
// ---------------------------------------------------------------------------
__global__ __launch_bounds__(32) void pam_combine(
    const float* __restrict__ part, const float* __restrict__ x,
    const float* __restrict__ alphap, float* __restrict__ out, int nsplit) {
  int tile  = blockIdx.x;
  int b     = tile >> 8;
  int qbase = (tile & 255) << 4;
  int lane  = threadIdx.x;
  int lq    = lane & 15;
  int g     = lane >> 4;
  const float* pb = part + (size_t)tile * nsplit * PART_F;

  float w[8];
  float M = -1e30f;
  for (int s = 0; s < nsplit; ++s) M = fmaxf(M, pb[s * PART_F + 1024 + lane]);
  float L = 0.f;
  for (int s = 0; s < nsplit; ++s) {
    w[s] = EXP2((pb[s * PART_F + 1024 + lane] - M) * L2E);
    L += pb[s * PART_F + 1056 + lane] * w[s];
  }
  L += __shfl_xor(L, 16, 32);
  float inv = (*alphap) / L;
#pragma unroll
  for (int cs = 0; cs < 4; ++cs)
#pragma unroll
    for (int r = 0; r < 8; ++r) {
      float a = 0.f;
      for (int s = 0; s < nsplit; ++s)
        a += pb[s * PART_F + (cs * 8 + r) * 32 + lane] * w[s];
      int c = cs * 16 + r + 8 * g;
      size_t idx = ((size_t)b * NC + c) * NN + qbase + lq;
      out[idx] = fmaf(a, inv, x[idx]);
    }
}

// ---------------------------------------------------------------------------
extern "C" void kernel_launch(void* const* d_in, const int* in_sizes, int n_in,
                              void* d_out, int out_size, void* d_ws, size_t ws_size,
                              hipStream_t stream) {
  const float* x      = (const float*)d_in[0];
  const float* Wq     = (const float*)d_in[1];
  const float* bq     = (const float*)d_in[2];
  const float* Wk     = (const float*)d_in[3];
  const float* bk     = (const float*)d_in[4];
  const float* Wv     = (const float*)d_in[5];
  const float* bv     = (const float*)d_in[6];
  const float* alphap = (const float*)d_in[7];

  float* out = (float*)d_out;                       // [4,64,64,64]
  float* qk  = out + (size_t)NB * NC * NN;          // [4,4096,4096]

  char* ws = (char*)d_ws;
  __bf16* Qt = (__bf16*)(ws);                                  // 256 KB
  __bf16* Kt = (__bf16*)(ws + 262144);                         // 256 KB
  __bf16* Vm = (__bf16*)(ws + 524288);                         // 2 MB
  char*   zp = ws + 2621440;                                   // 256 B zeros
  float*  part = (float*)(ws + 2621696);
  const size_t base = 2621696;
  const size_t per  = (size_t)1024 * PART_F * sizeof(float);   // ~4.35 MB

  int nsplit = 1, final_out = 1;                    // deterministic in ws_size
  if (ws_size >= base + 4 * per)      { nsplit = 4; final_out = 0; }
  else if (ws_size >= base + 2 * per) { nsplit = 2; final_out = 0; }

  pam_proj<<<dim3(NN / 256, NB), 256, 0, stream>>>(x, Wq, bq, Wk, bk, Wv, bv,
                                                   Qt, Kt, Vm, (unsigned*)zp);
  pam_attn<<<dim3(1024 * nsplit), 32, 0, stream>>>(
      Qt, Kt, Vm, (const __bf16*)zp, x, alphap, qk, out, part, nsplit,
      final_out);
  if (!final_out)
    pam_combine<<<dim3(1024), 32, 0, stream>>>(part, x, alphap, out, nsplit);
}
